// CausalSelfAttention_29429115912196
// MI455X (gfx1250) — matrix operand order repaired and checked by reading, not compiled, not run on hardware
//
#include <hip/hip_runtime.h>
#include <hip/hip_bf16.h>

// ---------------------------------------------------------------------------
// CDNA5 (gfx1250) causal self-attention forward, bf16 WMMA datapath.
// wave32; v_wmma_f32_16x16x32_bf16 for all four matmuls.
// Async global->LDS staging (ASYNCcnt) for the GEMM A tiles.
// ---------------------------------------------------------------------------

typedef __attribute__((ext_vector_type(16))) __bf16 v16bf;
typedef __attribute__((ext_vector_type(8)))  float  v8f;

union FragA { v16bf v; unsigned int u[8]; };
union AccF  { v8f  v; float f[8]; };

__device__ __forceinline__ unsigned short f2bf(float f) {
  union { float f; unsigned int u; } c; c.f = f;
  unsigned int r = c.u + 0x7FFFu + ((c.u >> 16) & 1u);  // round-to-nearest-even
  return (unsigned short)(r >> 16);
}
__device__ __forceinline__ float bf2f(unsigned short b) {
  union { unsigned int u; float f; } c; c.u = ((unsigned int)b) << 16;
  return c.f;
}

// ---- CDNA5 async global->LDS copy (tracked by ASYNCcnt) -------------------
// Generic pointers to LDS on AMDGPU carry the raw LDS byte offset in their
// low 32 bits (aperture lives in the high half), so VDST gets that offset.
__device__ __forceinline__ void async_copy_b128(void* lds_ptr, const void* gptr) {
  unsigned lds_off = (unsigned)(unsigned long long)lds_ptr;
  unsigned long long ga = (unsigned long long)gptr;
  asm volatile("global_load_async_to_lds_b128 %0, %1, off"
               :: "v"(lds_off), "v"(ga) : "memory");
}
__device__ __forceinline__ void wait_asynccnt0() {
  asm volatile("s_wait_asynccnt 0x0" ::: "memory");
}

// ---------------------------------------------------------------------------
// Kernel 1: fp32 -> bf16 conversion
// ---------------------------------------------------------------------------
__global__ void cvt_f32_to_bf16(const float* __restrict__ in,
                                unsigned short* __restrict__ out, int n) {
  int i = blockIdx.x * blockDim.x + threadIdx.x;
  if (i < n) out[i] = f2bf(in[i]);
}

// ---------------------------------------------------------------------------
// Kernel 2/5: LDS-tiled bf16 GEMM, fp32 accum via v_wmma_f32_16x16x32_bf16.
//   out[M,N] = A[M,K] @ W[K,N] + bias[N]
// Workgroup: 256 threads (8 waves), 128x128 tile. Wave (wm,wn) in 2x4 grid
// computes a 64x32 sub-tile = 4 M-tiles x 2 N-tiles, 8 WMMAs per K-step.
// A tile staged with async B128 copies; B tile transposed through VGPRs.
// ---------------------------------------------------------------------------
#define GT_M 128
#define GT_N 128
#define GT_K 32

__global__ __launch_bounds__(256)
void gemm_bf16_wmma(const unsigned short* __restrict__ A,
                    const unsigned short* __restrict__ W,
                    const float* __restrict__ bias,
                    unsigned short* __restrict__ outb,
                    float* __restrict__ outf,
                    int M, int N, int K, int out_is_f32) {
  __shared__ __align__(16) unsigned short As[GT_M][GT_K];      // 8 KB
  __shared__ __align__(16) unsigned short BsT[GT_N][GT_K + 4]; // 9 KB, transposed [n][k]

  const int tid  = threadIdx.x;
  const int lane = tid & 31;
  const int w    = tid >> 5;       // wave 0..7
  const int wm   = w >> 2;         // 0..1 -> 64-row band
  const int wn   = w & 3;          // 0..3 -> 32-col band
  const int lo   = lane & 15;
  const int hi   = lane >> 4;      // 0 or 1
  const int mbase = blockIdx.y * GT_M;
  const int nbase = blockIdx.x * GT_N;

  AccF acc[4][2];
#pragma unroll
  for (int mt = 0; mt < 4; ++mt)
#pragma unroll
    for (int nt = 0; nt < 2; ++nt)
#pragma unroll
      for (int i = 0; i < 8; ++i) acc[mt][nt].f[i] = 0.0f;

  // per-thread staging coordinates
  const int arow0 = (tid * 2) >> 2;            // A chunk 0 row
  const int acol0 = ((tid * 2) & 3) * 8;       // A chunk 0 col
  const int arow1 = (tid * 2 + 1) >> 2;
  const int acol1 = ((tid * 2 + 1) & 3) * 8;
  const int krow  = tid >> 3;                  // B: 0..31
  const int nc    = (tid & 7) * 16;            // B: 0..112

  for (int k0 = 0; k0 < K; k0 += GT_K) {
    // ---- A tile: 128x32 bf16 via async global->LDS B128 copies (ASYNCcnt)
    async_copy_b128(&As[arow0][acol0],
                    &A[(size_t)(mbase + arow0) * K + k0 + acol0]);
    async_copy_b128(&As[arow1][acol1],
                    &A[(size_t)(mbase + arow1) * K + k0 + acol1]);

    // ---- B tile transposed: 32(K) x 128(N) -> BsT[n][k]
    {
      const unsigned short* src = &W[(size_t)(k0 + krow) * N + nbase + nc];
      unsigned short tmp[16];
      *(int4*)&tmp[0] = *(const int4*)src;
      *(int4*)&tmp[8] = *(const int4*)(src + 8);
#pragma unroll
      for (int j = 0; j < 16; ++j) BsT[nc + j][krow] = tmp[j];
    }
    if (k0 + GT_K < K) {  // hint next W tile into cache (global_prefetch_b8)
      __builtin_prefetch(&W[(size_t)(k0 + GT_K + krow) * N + nbase + nc], 0, 1);
    }
    wait_asynccnt0();     // our async A copies landed in LDS
    __syncthreads();

    // ---- build fragments per ISA 16-bit layouts
    FragA af[4], bf[2];
#pragma unroll
    for (int mt = 0; mt < 4; ++mt) {
      int mrow = wm * 64 + mt * 16 + lo;
#pragma unroll
      for (int i = 0; i < 8; ++i) {
        int kp = 2 * i + ((i >= 4) ? 8 : 0) + hi * 8;   // A: K pair base
        af[mt].u[i] = *(const unsigned int*)&As[mrow][kp];
      }
    }
#pragma unroll
    for (int nt = 0; nt < 2; ++nt) {
      int ncol = wn * 32 + nt * 16 + lo;
#pragma unroll
      for (int i = 0; i < 8; ++i) {
        int kp = 2 * i + ((i >= 4) ? 8 : 0) + hi * 8;   // B: K pair base
        bf[nt].u[i] = *(const unsigned int*)&BsT[ncol][kp];
      }
    }
#pragma unroll
    for (int mt = 0; mt < 4; ++mt)
#pragma unroll
      for (int nt = 0; nt < 2; ++nt)
        acc[mt][nt].v = __builtin_amdgcn_wmma_f32_16x16x32_bf16(
            false, af[mt].v, false, bf[nt].v, (short)0, acc[mt][nt].v,
            false, false);
    __syncthreads();
  }

  // ---- epilogue: C/D layout M = r + hi*8, N = lo
#pragma unroll
  for (int nt = 0; nt < 2; ++nt) {
    int n  = nbase + wn * 32 + nt * 16 + lo;
    float bv = bias[n];
#pragma unroll
    for (int mt = 0; mt < 4; ++mt) {
#pragma unroll
      for (int r = 0; r < 8; ++r) {
        int m = mbase + wm * 64 + mt * 16 + r + hi * 8;
        float val = acc[mt][nt].f[r] + bv;
        if (out_is_f32) outf[(size_t)m * N + n] = val;
        else            outb[(size_t)m * N + n] = f2bf(val);
      }
    }
  }
}

// ---------------------------------------------------------------------------
// Kernel 3: RoPE (full rope, D=64, pairs (d, d+32)) + head split.
//   qkv bf16 [B,T,3C] -> qh,kh bf16 [B*H,T,D]; v -> vT bf16 [B*H,D,T]
// ---------------------------------------------------------------------------
__global__ void rope_split(const unsigned short* __restrict__ qkv,
                           unsigned short* __restrict__ qh,
                           unsigned short* __restrict__ kh,
                           unsigned short* __restrict__ vT, int T) {
  const int C = 1024, D = 64;
  int idx = blockIdx.x * blockDim.x + threadIdx.x;
  int total = 2 * T * 16 * 32;
  if (idx >= total) return;
  int dp = idx & 31;
  int tmp = idx >> 5;
  int h = tmp & 15; tmp >>= 4;
  int t = tmp % T;
  int b = tmp / T;

  size_t base = (size_t)(b * T + t) * (3 * C);
  float inv_ts = __powf(10000.0f, -(float)dp / 32.0f);
  float ang = (float)t * inv_ts;
  float sn, cs;
  __sincosf(ang, &sn, &cs);

  int c0 = h * D + dp, c1 = c0 + 32;
  float q0 = bf2f(qkv[base + c0]),     q1 = bf2f(qkv[base + c1]);
  float k0 = bf2f(qkv[base + C + c0]), k1 = bf2f(qkv[base + C + c1]);

  size_t ho = ((size_t)(b * 16 + h) * T + t) * D;
  qh[ho + dp]      = f2bf(q0 * cs - q1 * sn);
  qh[ho + dp + 32] = f2bf(q1 * cs + q0 * sn);
  kh[ho + dp]      = f2bf(k0 * cs - k1 * sn);
  kh[ho + dp + 32] = f2bf(k1 * cs + k0 * sn);

  size_t vb = (size_t)(b * 16 + h) * D * T;
  vT[vb + (size_t)dp * T + t]        = qkv[base + 2 * C + c0];
  vT[vb + (size_t)(dp + 32) * T + t] = qkv[base + 2 * C + c1];
}

// ---------------------------------------------------------------------------
// Kernel 4: causal flash attention. Block = 256 thr (8 waves); wave owns 16
// query rows; streams 32-key tiles: S = Q@K^T (4 WMMA), online softmax with
// shfl_xor row reductions, P re-layout via wave-private LDS, P@V (4 WMMA).
// ---------------------------------------------------------------------------
__global__ __launch_bounds__(256)
void flash_attn(const unsigned short* __restrict__ qh,
                const unsigned short* __restrict__ kh,
                const unsigned short* __restrict__ vT,
                unsigned short* __restrict__ yb, int T) {
  const int D = 64;
  __shared__ __align__(16) unsigned short Pl[8][16][32];  // 8 KB, per-wave patch

  const int tid  = threadIdx.x;
  const int lane = tid & 31;
  const int w    = tid >> 5;
  const int lo   = lane & 15;
  const int hi   = lane >> 4;
  const int bh   = blockIdx.y;            // b*16 + h
  const int b    = bh >> 4;
  const int h    = bh & 15;
  const int qbase = blockIdx.x * 128 + w * 16;

  const unsigned short* q = qh + (size_t)bh * T * D;
  const unsigned short* k = kh + (size_t)bh * T * D;
  const unsigned short* v = vT + (size_t)bh * D * T;

  // Q fragments (d 0..31 and 32..63), loaded once, reused across key loop.
  FragA qa[2];
  {
    int qrow = qbase + lo;
#pragma unroll
    for (int f = 0; f < 2; ++f)
#pragma unroll
      for (int i = 0; i < 8; ++i) {
        int d = 2 * i + ((i >= 4) ? 8 : 0) + hi * 8 + f * 32;
        qa[f].u[i] = *(const unsigned int*)&q[(size_t)qrow * D + d];
      }
  }

  AccF o[4];
  float m[8], l[8];
#pragma unroll
  for (int nt = 0; nt < 4; ++nt)
#pragma unroll
    for (int i = 0; i < 8; ++i) o[nt].f[i] = 0.0f;
#pragma unroll
  for (int r = 0; r < 8; ++r) { m[r] = -1e30f; l[r] = 0.0f; }

  const float scale = 0.125f;  // 1/sqrt(64)
  const int kend = qbase + 16; // causal: keys < qbase+16

  for (int kb = 0; kb < kend; kb += 32) {
    // ---- S = Q @ K^T for 32 keys (two 16-key column tiles)
    AccF s[2];
#pragma unroll
    for (int nc = 0; nc < 2; ++nc) {
      FragA kf0, kf1;
      int key = kb + nc * 16 + lo;
#pragma unroll
      for (int i = 0; i < 8; ++i) {
        int d = 2 * i + ((i >= 4) ? 8 : 0) + hi * 8;
        kf0.u[i] = *(const unsigned int*)&k[(size_t)key * D + d];
        kf1.u[i] = *(const unsigned int*)&k[(size_t)key * D + d + 32];
      }
#pragma unroll
      for (int i = 0; i < 8; ++i) s[nc].f[i] = 0.0f;
      s[nc].v = __builtin_amdgcn_wmma_f32_16x16x32_bf16(
          false, qa[0].v, false, kf0.v, (short)0, s[nc].v, false, false);
      s[nc].v = __builtin_amdgcn_wmma_f32_16x16x32_bf16(
          false, qa[1].v, false, kf1.v, (short)0, s[nc].v, false, false);
    }

    // ---- scale + causal mask + per-row tile max
    float tmax[8];
#pragma unroll
    for (int r = 0; r < 8; ++r) {
      int qrow = qbase + r + hi * 8;
      float s0 = s[0].f[r] * scale;
      float s1 = s[1].f[r] * scale;
      if (kb + lo      > qrow) s0 = -1e30f;
      if (kb + 16 + lo > qrow) s1 = -1e30f;
      s[0].f[r] = s0; s[1].f[r] = s1;
      tmax[r] = fmaxf(s0, s1);
    }
#pragma unroll
    for (int r = 0; r < 8; ++r) {
      tmax[r] = fmaxf(tmax[r], __shfl_xor(tmax[r], 1, 32));
      tmax[r] = fmaxf(tmax[r], __shfl_xor(tmax[r], 2, 32));
      tmax[r] = fmaxf(tmax[r], __shfl_xor(tmax[r], 4, 32));
      tmax[r] = fmaxf(tmax[r], __shfl_xor(tmax[r], 8, 32));
    }

    // ---- online softmax update; stage P (bf16) into wave-private LDS
    float tsum[8];
#pragma unroll
    for (int r = 0; r < 8; ++r) {
      float mn    = fmaxf(m[r], tmax[r]);
      float alpha = __expf(m[r] - mn);
      m[r] = mn;
      float p0 = __expf(s[0].f[r] - mn);
      float p1 = __expf(s[1].f[r] - mn);
      tsum[r] = p0 + p1;
      l[r] *= alpha;
#pragma unroll
      for (int nt = 0; nt < 4; ++nt) o[nt].f[r] *= alpha;
      int row = r + hi * 8;
      Pl[w][row][lo]      = f2bf(p0);
      Pl[w][row][lo + 16] = f2bf(p1);
    }
#pragma unroll
    for (int r = 0; r < 8; ++r) {
      tsum[r] += __shfl_xor(tsum[r], 1, 32);
      tsum[r] += __shfl_xor(tsum[r], 2, 32);
      tsum[r] += __shfl_xor(tsum[r], 4, 32);
      tsum[r] += __shfl_xor(tsum[r], 8, 32);
      l[r] += tsum[r];
    }

    // ---- P fragment (A, 16x32) from LDS; same-wave LDS ops are in-order.
    FragA pf;
#pragma unroll
    for (int i = 0; i < 8; ++i) {
      int kp = 2 * i + ((i >= 4) ? 8 : 0) + hi * 8;
      pf.u[i] = *(const unsigned int*)&Pl[w][lo][kp];
    }

    // ---- O += P @ V (V from transposed [D,T] layout -> contiguous pairs)
#pragma unroll
    for (int nt = 0; nt < 4; ++nt) {
      FragA vf;
      int d = nt * 16 + lo;
#pragma unroll
      for (int i = 0; i < 8; ++i) {
        int koff = 2 * i + ((i >= 4) ? 8 : 0) + hi * 8;
        vf.u[i] = *(const unsigned int*)&v[(size_t)d * T + kb + koff];
      }
      o[nt].v = __builtin_amdgcn_wmma_f32_16x16x32_bf16(
          false, pf.v, false, vf.v, (short)0, o[nt].v, false, false);
    }
  }

  // ---- normalize and write y (bf16, [B,T,C] with c = h*64 + d)
#pragma unroll
  for (int nt = 0; nt < 4; ++nt) {
    int c = h * 64 + nt * 16 + lo;
#pragma unroll
    for (int r = 0; r < 8; ++r) {
      int t = qbase + r + hi * 8;
      float val = o[nt].f[r] / l[r];
      yb[((size_t)b * T + t) * 1024 + c] = f2bf(val);
    }
  }
}

// ---------------------------------------------------------------------------
// Host-side launch
// ---------------------------------------------------------------------------
extern "C" void kernel_launch(void* const* d_in, const int* in_sizes, int n_in,
                              void* d_out, int out_size, void* d_ws, size_t ws_size,
                              hipStream_t stream) {
  (void)in_sizes; (void)n_in; (void)out_size; (void)ws_size;
  const float* x      = (const float*)d_in[0];   // [2,2048,1024]
  const float* W_attn = (const float*)d_in[1];   // [1024,3072]
  const float* b_attn = (const float*)d_in[2];   // [3072]
  const float* W_proj = (const float*)d_in[3];   // [1024,1024]
  const float* b_proj = (const float*)d_in[4];   // [1024]
  float* out = (float*)d_out;

  const int B = 2, T = 2048, C = 1024;
  const int M = B * T;           // 4096
  const int N3 = 3 * C;          // 3072

  // workspace carving (bf16 buffers, 256B aligned)
  char* ws = (char*)d_ws;
  size_t off = 0;
  auto carve = [&](size_t elems) {
    void* p = ws + off;
    off = (off + elems * 2 + 255) & ~(size_t)255;
    return (unsigned short*)p;
  };
  unsigned short* xb   = carve((size_t)M * C);        // x bf16
  unsigned short* wAb  = carve((size_t)C * N3);       // W_attn bf16
  unsigned short* wPb  = carve((size_t)C * C);        // W_proj bf16
  unsigned short* qkvb = carve((size_t)M * N3);       // qkv bf16
  unsigned short* qhb  = carve((size_t)M * C);        // [B*H,T,D]
  unsigned short* khb  = carve((size_t)M * C);
  unsigned short* vTb  = carve((size_t)M * C);        // [B*H,D,T]
  unsigned short* ybb  = carve((size_t)M * C);        // attention out bf16

  // 1) conversions
  {
    int n;
    n = M * C;  cvt_f32_to_bf16<<<(n + 255) / 256, 256, 0, stream>>>(x, xb, n);
    n = C * N3; cvt_f32_to_bf16<<<(n + 255) / 256, 256, 0, stream>>>(W_attn, wAb, n);
    n = C * C;  cvt_f32_to_bf16<<<(n + 255) / 256, 256, 0, stream>>>(W_proj, wPb, n);
  }
  // 2) QKV projection
  {
    dim3 g(N3 / GT_N, M / GT_M);
    gemm_bf16_wmma<<<g, 256, 0, stream>>>(xb, wAb, b_attn, qkvb, nullptr,
                                          M, N3, C, 0);
  }
  // 3) RoPE + head split (+ V transpose)
  {
    int n = B * T * 16 * 32;
    rope_split<<<(n + 255) / 256, 256, 0, stream>>>(qkvb, qhb, khb, vTb, T);
  }
  // 4) flash attention
  {
    dim3 g(T / 128, B * 16);
    flash_attn<<<g, 256, 0, stream>>>(qhb, khb, vTb, ybb, T);
  }
  // 5) output projection -> fp32 d_out
  {
    dim3 g(C / GT_N, M / GT_M);
    gemm_bf16_wmma<<<g, 256, 0, stream>>>(ybb, wPb, b_proj, nullptr, out,
                                          M, C, C, 1);
  }
}
